// MaskedSelfAttention_47682726920349
// MI455X (gfx1250) — matrix operand order repaired and checked by reading, not compiled, not run on hardware
//
#include <hip/hip_runtime.h>

#define BATCH 16
#define SEQ   1024
#define EMBED 1024
#define IMG   256
#define SCALE 0.03125f  // 1/sqrt(1024)

typedef unsigned short u16;
typedef __attribute__((ext_vector_type(16))) __bf16          v16bf;
typedef __attribute__((ext_vector_type(16))) unsigned short  v16u;
typedef __attribute__((ext_vector_type(8)))  float           v8f;

union FragU {
  uint4 q[2];
  v16u  us;
  v16bf bf;
};

__device__ __forceinline__ u16 f32_to_bf16(float f) {
  unsigned int u = __float_as_uint(f);
  u += 0x7fffu + ((u >> 16) & 1u);   // round-to-nearest-even
  return (u16)(u >> 16);
}

__device__ __forceinline__ v8f wmma_bf16(v16bf a, v16bf b, v8f c) {
  return __builtin_amdgcn_wmma_f32_16x16x32_bf16(
      /*neg_a=*/false, a, /*neg_b=*/false, b,
      /*c_mod=*/(short)0, c, /*reuse_a=*/false, /*reuse_b=*/false);
}

// A fragment: 16(M) x 32(K) bf16 tile; memory row-major [M][K], base at (m0,k0).
// lane L: M = L%16; element j -> K = (j/8)*16 + (L/16)*8 + (j%8)  (ISA 7.12.2)
__device__ __forceinline__ v16bf load_a_frag(const u16* base, int stride, int lane) {
  const u16* p = base + (size_t)(lane & 15) * stride + ((lane >> 4) << 3);
  FragU f;
  f.q[0] = *(const uint4*)p;
  f.q[1] = *(const uint4*)(p + 16);
  return f.bf;
}

// B fragment: 32(K) x 16(N); memory layout [N][K] with K contiguous.
// lane L: N = L%16; element j -> K = (j/8)*16 + (L/16)*8 + (j%8)
__device__ __forceinline__ v16bf load_b_frag_kmajor(const u16* base, int stride, int lane) {
  const u16* p = base + (size_t)(lane & 15) * stride + ((lane >> 4) << 3);
  FragU f;
  f.q[0] = *(const uint4*)p;
  f.q[1] = *(const uint4*)(p + 16);
  return f.bf;
}

// ---------------------------------------------------------------- fp32 -> bf16
__global__ void cvt_bf16_kernel(const float* __restrict__ src,
                                u16* __restrict__ dst, int n) {
  int i = blockIdx.x * blockDim.x + threadIdx.x;
  int stride = gridDim.x * blockDim.x;
  for (; i < n; i += stride) dst[i] = f32_to_bf16(src[i]);
}

// ------------------------------------------- O[m,d] = sum_e X[m,e]*W[d,e] + b[d]
// X: [16384, 1024] bf16, W: [1024, 1024] bf16 (row d, col e).
// One wave -> 32x64 output tile (8 WMMAs per 32-wide K-step).
// transpose_out=false: O[row*EMBED + col] (Q, K)
// transpose_out=true : O[b*EMBED*SEQ + col*SEQ + s] i.e. V^T [B][E][S]
__global__ void __launch_bounds__(256)
qkv_gemm_kernel(const u16* __restrict__ X, const u16* __restrict__ W,
                const float* __restrict__ bias, u16* __restrict__ O,
                int transpose_out) {
  const int wave = threadIdx.x >> 5, lane = threadIdx.x & 31;
  const int t  = blockIdx.x * 8 + wave;      // 8192 wave tiles
  const int m0 = (t >> 4) << 5;              // 512 M-tiles of 32
  const int n0 = (t & 15) << 6;              // 16 N-tiles of 64

  const u16* Xa0 = X + (size_t)m0 * EMBED;
  const u16* Xa1 = X + (size_t)(m0 + 16) * EMBED;
  // prefetch helpers: 32 lanes cover the tile's rows, one slab (512 elems) ahead
  const u16* pfA  = X + (size_t)(m0 + lane) * EMBED;
  const u16* pfB0 = W + (size_t)(n0 + lane) * EMBED;
  const u16* pfB1 = W + (size_t)(n0 + 32 + lane) * EMBED;

  v8f acc[8] = {};
  for (int e0 = 0; e0 < EMBED; e0 += 32) {
    __builtin_prefetch(pfA + e0 + 512, 0, 3);
    __builtin_prefetch(pfB0 + e0 + 512, 0, 3);
    __builtin_prefetch(pfB1 + e0 + 512, 0, 3);
    // all fragment loads issued before the WMMA burst
    v16bf a0 = load_a_frag(Xa0 + e0, EMBED, lane);
    v16bf a1 = load_a_frag(Xa1 + e0, EMBED, lane);
    v16bf b0 = load_b_frag_kmajor(W + (size_t)(n0     ) * EMBED + e0, EMBED, lane);
    v16bf b1 = load_b_frag_kmajor(W + (size_t)(n0 + 16) * EMBED + e0, EMBED, lane);
    v16bf b2 = load_b_frag_kmajor(W + (size_t)(n0 + 32) * EMBED + e0, EMBED, lane);
    v16bf b3 = load_b_frag_kmajor(W + (size_t)(n0 + 48) * EMBED + e0, EMBED, lane);
    acc[0] = wmma_bf16(a0, b0, acc[0]);
    acc[4] = wmma_bf16(a1, b0, acc[4]);
    acc[1] = wmma_bf16(a0, b1, acc[1]);
    acc[5] = wmma_bf16(a1, b1, acc[5]);
    acc[2] = wmma_bf16(a0, b2, acc[2]);
    acc[6] = wmma_bf16(a1, b2, acc[6]);
    acc[3] = wmma_bf16(a0, b3, acc[3]);
    acc[7] = wmma_bf16(a1, b3, acc[7]);
  }

  const int rbase = (lane >> 4) << 3;        // C layout: row = r + 8*(lane/16)
#pragma unroll
  for (int h = 0; h < 2; ++h) {              // two 16-row halves
    const int mrow = m0 + h * 16 + rbase;
#pragma unroll
    for (int n = 0; n < 4; ++n) {
      const int col = n0 + n * 16 + (lane & 15);
      const float bv = bias[col];
      if (!transpose_out) {
#pragma unroll
        for (int r = 0; r < 8; ++r)
          O[(size_t)(mrow + r) * EMBED + col] = f32_to_bf16(acc[h * 4 + n][r] + bv);
      } else {
#pragma unroll
        for (int r = 0; r < 8; ++r) {
          const int row = mrow + r;
          const int bb = row >> 10, s = row & (SEQ - 1);
          O[(size_t)bb * EMBED * SEQ + (size_t)col * SEQ + s] =
              f32_to_bf16(acc[h * 4 + n][r] + bv);
        }
      }
    }
  }
}

// --------------------------------------------------- flash attention, bf16 WMMA
// grid: (SEQ/32, BATCH), block 256 (8 waves). 32 queries per block.
// Per 32-key block: waves split d=1024 for QK^T (16 WMMAs each), LDS reduce of
// the 32x32 score tile, online softmax by 32 row-threads, then each wave does
// PV for its 32x64 output slice (8 WMMAs, V^T gives contiguous B-fragments).
__global__ void __launch_bounds__(256)
attn_kernel(const u16* __restrict__ Q, const u16* __restrict__ K,
            const u16* __restrict__ Vt, float* __restrict__ O) {
  __shared__ alignas(16) float sS[8][1024];  // per-wave partial 32x32 scores
  __shared__ alignas(16) float sRed[1024];   // reduced/masked scores
  __shared__ alignas(32) u16   sP[1024];     // 32x32 probabilities (bf16)
  __shared__ float sM[32], sL[32], sAlpha[32];

  const int tid = threadIdx.x, wave = tid >> 5, lane = tid & 31;
  const int b = blockIdx.y, q0 = blockIdx.x << 5;
  const u16* Qb  = Q  + (size_t)b * SEQ * EMBED;
  const u16* Kb  = K  + (size_t)b * SEQ * EMBED;
  const u16* VtB = Vt + (size_t)b * EMBED * SEQ;
  float*     Ob  = O  + (size_t)b * SEQ * EMBED;

  if (tid < 32) { sM[tid] = -3.0e38f; sL[tid] = 0.0f; }
  __syncthreads();

  const int rbase = (lane >> 4) << 3;
  const int dw = wave << 6;                  // this wave's 64-wide output slice
  const int ds = wave << 7;                  // this wave's 128-wide score d-slice
  v8f acc[8] = {};                           // acc[h*4+n]: rows h*16+, cols n*16+

  const int kmax = (q0 + 32 > IMG) ? (q0 + 32) : IMG;   // exclusive key bound
  const int nkb  = (kmax + 31) >> 5;

  for (int kb = 0; kb < nkb; ++kb) {
    const int k0 = kb << 5;
    // ---- partial scores: this wave covers d in [ds, ds+128)
    v8f sc[4] = {};                          // sc[h*2+n]
#pragma unroll
    for (int dd = 0; dd < 4; ++dd) {
      const int d0 = ds + dd * 32;
      v16bf a0 = load_a_frag(Qb + (size_t)q0 * EMBED + d0, EMBED, lane);
      v16bf a1 = load_a_frag(Qb + (size_t)(q0 + 16) * EMBED + d0, EMBED, lane);
      v16bf b0 = load_b_frag_kmajor(Kb + (size_t)k0 * EMBED + d0, EMBED, lane);
      v16bf b1 = load_b_frag_kmajor(Kb + (size_t)(k0 + 16) * EMBED + d0, EMBED, lane);
      sc[0] = wmma_bf16(a0, b0, sc[0]);
      sc[2] = wmma_bf16(a1, b0, sc[2]);
      sc[1] = wmma_bf16(a0, b1, sc[1]);
      sc[3] = wmma_bf16(a1, b1, sc[3]);
    }
#pragma unroll
    for (int h = 0; h < 2; ++h)
#pragma unroll
      for (int r = 0; r < 8; ++r) {
        const int row = h * 16 + rbase + r;
        sS[wave][row * 32 + (lane & 15)]      = sc[h * 2 + 0][r];
        sS[wave][row * 32 + 16 + (lane & 15)] = sc[h * 2 + 1][r];
      }
    __syncthreads();
    // ---- reduce 8 partials, scale, apply prefix-causal mask
    for (int idx = tid; idx < 1024; idx += 256) {
      float s = 0.0f;
#pragma unroll
      for (int w = 0; w < 8; ++w) s += sS[w][idx];
      const int i = q0 + (idx >> 5), j = k0 + (idx & 31);
      const bool vis = (j < IMG) || (j <= i);
      sRed[idx] = vis ? s * SCALE : -1.0e9f;
    }
    __syncthreads();
    // ---- online softmax update (one thread per query row)
    if (tid < 32) {
      const float mo = sM[tid];
      float mm = mo;
#pragma unroll
      for (int c = 0; c < 32; ++c) { float v = sRed[tid * 32 + c]; mm = v > mm ? v : mm; }
      const float alpha = __expf(mo - mm);
      float ls = 0.0f;
#pragma unroll
      for (int c = 0; c < 32; ++c) {
        const float p = __expf(sRed[tid * 32 + c] - mm);
        sP[tid * 32 + c] = f32_to_bf16(p);
        ls += p;
      }
      sL[tid] = sL[tid] * alpha + ls;
      sM[tid] = mm;
      sAlpha[tid] = alpha;
    }
    __syncthreads();
    // ---- rescale accumulators, then PV (V B-fragment shared by both P halves)
    float ar[2][8];
#pragma unroll
    for (int h = 0; h < 2; ++h)
#pragma unroll
      for (int r = 0; r < 8; ++r) ar[h][r] = sAlpha[h * 16 + rbase + r];
#pragma unroll
    for (int h = 0; h < 2; ++h)
#pragma unroll
      for (int n = 0; n < 4; ++n)
#pragma unroll
        for (int r = 0; r < 8; ++r) acc[h * 4 + n][r] *= ar[h][r];

    v16bf p0 = load_a_frag((const u16*)sP, 32, lane);            // P rows 0..15
    v16bf p1 = load_a_frag((const u16*)sP + 16 * 32, 32, lane);  // P rows 16..31
#pragma unroll
    for (int n = 0; n < 4; ++n) {
      v16bf bv = load_b_frag_kmajor(VtB + (size_t)(dw + n * 16) * SEQ + k0, SEQ, lane);
      acc[n]     = wmma_bf16(p0, bv, acc[n]);
      acc[4 + n] = wmma_bf16(p1, bv, acc[4 + n]);
    }
    __syncthreads();
  }
  // ---- normalize and store fp32 output
  float linv[2][8];
#pragma unroll
  for (int h = 0; h < 2; ++h)
#pragma unroll
    for (int r = 0; r < 8; ++r) linv[h][r] = 1.0f / sL[h * 16 + rbase + r];
#pragma unroll
  for (int h = 0; h < 2; ++h)
#pragma unroll
    for (int n = 0; n < 4; ++n) {
      const int col = dw + n * 16 + (lane & 15);
#pragma unroll
      for (int r = 0; r < 8; ++r) {
        const int row = q0 + h * 16 + rbase + r;
        Ob[(size_t)row * EMBED + col] = acc[h * 4 + n][r] * linv[h][r];
      }
    }
}

extern "C" void kernel_launch(void* const* d_in, const int* in_sizes, int n_in,
                              void* d_out, int out_size, void* d_ws, size_t ws_size,
                              hipStream_t stream) {
  (void)in_sizes; (void)n_in; (void)out_size; (void)ws_size;
  const float* x  = (const float*)d_in[0];
  const float* Wq = (const float*)d_in[1];
  const float* bq = (const float*)d_in[2];
  const float* Wk = (const float*)d_in[3];
  const float* bk = (const float*)d_in[4];
  const float* Wv = (const float*)d_in[5];
  const float* bv = (const float*)d_in[6];
  float* out = (float*)d_out;

  const size_t nx = (size_t)BATCH * SEQ * EMBED;  // 16 Mi elements
  const size_t nw = (size_t)EMBED * EMBED;        // 1 Mi elements
  u16* xbf = (u16*)d_ws;
  u16* qbf = xbf + nx;
  u16* kbf = qbf + nx;
  u16* vtb = kbf + nx;      // V^T: [B][EMBED][SEQ]
  u16* wqb = vtb + nx;
  u16* wkb = wqb + nw;
  u16* wvb = wkb + nw;

  cvt_bf16_kernel<<<4096, 256, 0, stream>>>(x,  xbf, (int)nx);
  cvt_bf16_kernel<<<512,  256, 0, stream>>>(Wq, wqb, (int)nw);
  cvt_bf16_kernel<<<512,  256, 0, stream>>>(Wk, wkb, (int)nw);
  cvt_bf16_kernel<<<512,  256, 0, stream>>>(Wv, wvb, (int)nw);

  qkv_gemm_kernel<<<1024, 256, 0, stream>>>(xbf, wqb, bq, qbf, 0);
  qkv_gemm_kernel<<<1024, 256, 0, stream>>>(xbf, wkb, bk, kbf, 0);
  qkv_gemm_kernel<<<1024, 256, 0, stream>>>(xbf, wvb, bv, vtb, 1);

  attn_kernel<<<dim3(SEQ / 32, BATCH), 256, 0, stream>>>(qbf, kbf, vtb, out);
}
